// T5Gemma2Decoder_61675730371280
// MI455X (gfx1250) — matrix operand order repaired and checked by reading, not compiled, not run on hardware
//
#include <hip/hip_runtime.h>
#include <hip/hip_bf16.h>
#include <stdint.h>

#define B_  2
#define S_  1024
#define E_  1024
#define D_  2048
#define H_  8
#define KV_ 4
#define HD_ 256
#define L_  (S_ + E_)
#define WIN_ 512

typedef __attribute__((ext_vector_type(16))) __bf16 v16bf;
typedef __attribute__((ext_vector_type(8)))  __bf16 v8bf;
typedef __attribute__((ext_vector_type(4)))  __bf16 v4bf;
typedef __attribute__((ext_vector_type(8)))  float  v8f;
typedef __attribute__((ext_vector_type(8)))  unsigned short u16x8;

__device__ __forceinline__ __bf16 f2bf(float f){
  unsigned u = __float_as_uint(f);
  unsigned r = u + 0x7FFFu + ((u >> 16) & 1u);   // round-to-nearest-even
  unsigned short us = (unsigned short)(r >> 16);
  return __builtin_bit_cast(__bf16, us);
}

__device__ __forceinline__ v16bf cat8(v8bf a, v8bf b){
  return __builtin_shufflevector(a, b, 0,1,2,3,4,5,6,7,8,9,10,11,12,13,14,15);
}

__device__ __forceinline__ v8f wmma_bf16(v16bf a, v16bf b, v8f c){
  return __builtin_amdgcn_wmma_f32_16x16x32_bf16(false, a, false, b, (short)0, c,
                                                 false, false);
}

// ---------------- elementwise f32 -> bf16 ----------------
__global__ void k_cvt_bf16(const float* __restrict__ src, __bf16* __restrict__ dst, int n4){
  int i = blockIdx.x * blockDim.x + threadIdx.x;
  if (i < n4){
    const float4 v = reinterpret_cast<const float4*>(src)[i];
    v4bf o = { f2bf(v.x), f2bf(v.y), f2bf(v.z), f2bf(v.w) };
    reinterpret_cast<v4bf*>(dst)[i] = o;
  }
}

// ---------------- segment ids from position resets ----------------
__global__ void k_seg(const int* __restrict__ pos, int* __restrict__ seg){
  int b = blockIdx.x;
  if (threadIdx.x == 0){
    int cur = 0;
    for (int i = 0; i < S_; ++i){
      bool reset = (i == 0) || (pos[b*S_ + i] <= pos[b*S_ + i - 1]);
      if (reset) cur++;
      seg[b*S_ + i] = cur;
    }
  }
}

// ------------- RMS norm (+optional RoPE) + bf16 relayout -------------
// in:  (B, rows, nh, HD) f32   out: (B, nh, Lout, HD) bf16 at seq offset
__global__ __launch_bounds__(HD_) void k_norm_rope(
    const float* __restrict__ in, int rows, int nh,
    const float* __restrict__ normw,
    const float* __restrict__ cosp, const float* __restrict__ sinp,
    __bf16* __restrict__ out, int Lout, int seqoff)
{
  __shared__ float part[HD_/32];
  __shared__ float ybuf[HD_];
  __shared__ float sscale;
  int idx = blockIdx.x;
  int hh = idx % nh;
  int s  = (idx / nh) % rows;
  int b  = idx / (nh * rows);
  int d  = threadIdx.x;
  float x = in[(((size_t)b*rows + s)*nh + hh)*HD_ + d];
  float y = x;
  if (normw){
    float v = x * x;
    #pragma unroll
    for (int m = 16; m >= 1; m >>= 1) v += __shfl_xor(v, m, 32);
    if ((d & 31) == 0) part[d >> 5] = v;
    __syncthreads();
    if (d == 0){
      float ss = 0.f;
      #pragma unroll
      for (int i = 0; i < HD_/32; ++i) ss += part[i];
      sscale = rsqrtf(ss * (1.0f/HD_) + 1e-6f);
    }
    __syncthreads();
    y = x * sscale * (1.0f + normw[d]);
  }
  if (cosp){
    ybuf[d] = y;
    __syncthreads();
    float rot = (d < HD_/2) ? -ybuf[d + HD_/2] : ybuf[d - HD_/2];
    size_t ci = ((size_t)b*S_ + s)*HD_ + d;
    y = y * cosp[ci] + rot * sinp[ci];
  }
  out[(((size_t)b*nh + hh)*Lout + (seqoff + s))*HD_ + d] = f2bf(y);
}

// ---------------- bf16 WMMA GEMM: C(f32) = A(MxK) * B(KxN) ----------------
#define GTM 128
#define GTN 128
#define GTK 32
__global__ __launch_bounds__(256) void k_gemm_bf16(
    const unsigned short* __restrict__ A, const unsigned short* __restrict__ Bw,
    float* __restrict__ C, int M, int N, int K)
{
  __shared__ __align__(16) unsigned short sA[GTM*GTK];   // [m][k]
  __shared__ __align__(16) unsigned short sB[GTN*GTK];   // [n][k] (col-major)
  const int t  = threadIdx.x;
  const int bm = blockIdx.x * GTM;
  const int bn = blockIdx.y * GTN;
  const int w = t >> 5, lane = t & 31, hi = lane >> 4, ln = lane & 15;
  const int m0 = (w >> 1) * 32, n0 = (w & 1) * 64;
  v8f acc[2][4] = {};
  for (int k0 = 0; k0 < K; k0 += GTK){
    if (k0 + GTK < K)
      __builtin_prefetch(A + (size_t)(bm + (t >> 1))*K + k0 + GTK, 0, 1);
    #pragma unroll
    for (int it = 0; it < 2; ++it){                 // stage A (128x32)
      int i = t + it*256;
      int m = i >> 2, kq = (i & 3) * 8;
      *(u16x8*)(sA + m*GTK + kq) =
          *(const u16x8*)(A + (size_t)(bm + m)*K + k0 + kq);
    }
    #pragma unroll
    for (int it = 0; it < 2; ++it){                 // stage B (32x128 -> col major)
      int i = t + it*256;
      int k = i >> 4, nq = (i & 15) * 8;
      u16x8 vv = *(const u16x8*)(Bw + (size_t)(k0 + k)*N + bn + nq);
      #pragma unroll
      for (int j = 0; j < 8; ++j) sB[(nq + j)*GTK + k] = vv[j];
    }
    __syncthreads();
    v16bf af[2];
    #pragma unroll
    for (int mt = 0; mt < 2; ++mt){
      int row = m0 + mt*16 + ln;
      v8bf a0 = *(const v8bf*)(sA + row*GTK + 8*hi);
      v8bf a1 = *(const v8bf*)(sA + row*GTK + 16 + 8*hi);
      af[mt] = cat8(a0, a1);
    }
    #pragma unroll
    for (int nt = 0; nt < 4; ++nt){
      int col = n0 + nt*16 + ln;
      v8bf b0 = *(const v8bf*)(sB + col*GTK + 8*hi);
      v8bf b1 = *(const v8bf*)(sB + col*GTK + 16 + 8*hi);
      v16bf bfr = cat8(b0, b1);
      #pragma unroll
      for (int mt = 0; mt < 2; ++mt)
        acc[mt][nt] = wmma_bf16(af[mt], bfr, acc[mt][nt]);
    }
    __syncthreads();
  }
  #pragma unroll
  for (int mt = 0; mt < 2; ++mt)
    #pragma unroll
    for (int nt = 0; nt < 4; ++nt)
      #pragma unroll
      for (int r = 0; r < 8; ++r){
        int row = bm + m0 + mt*16 + r + 8*hi;
        int col = bn + n0 + nt*16 + ln;
        C[(size_t)row*N + col] = acc[mt][nt][r];
      }
}

// ---------------- flash attention with softcap + merged mask ----------------
__global__ __launch_bounds__(128) void k_attn(
    const unsigned short* __restrict__ qb,     // (B,H,S,HD) bf16
    const unsigned short* __restrict__ kall,   // (B,KV,L,HD) bf16
    const unsigned short* __restrict__ vall,   // (B,KV,L,HD) bf16
    const int* __restrict__ seg, const int* __restrict__ decm,
    const int* __restrict__ encm,
    __bf16* __restrict__ attnout)              // (B*S, H*HD) bf16
{
  __shared__ __align__(16) unsigned short sK[32*HD_];    // [key][d]
  __shared__ __align__(16) unsigned short sV[HD_*32];    // [d][key] (transposed)
  __shared__ __align__(16) __bf16 sP[4][16*32];          // per-wave probs
  const int tiles = S_ / 64;
  int blk = blockIdx.x;
  int b = blk / (H_ * tiles);
  int rem = blk % (H_ * tiles);
  int h = rem / tiles;
  int q0 = (rem % tiles) * 64;
  const int w = threadIdx.x >> 5, lane = threadIdx.x & 31;
  const int hi = lane >> 4, ln = lane & 15;
  const int q0w = q0 + w * 16;
  const int kvh = h / (H_ / KV_);
  const unsigned short* qbase = qb   + ((size_t)(b*H_  + h)   * S_) * HD_;
  const unsigned short* kbase = kall + ((size_t)(b*KV_ + kvh) * L_) * HD_;
  const unsigned short* vbase = vall + ((size_t)(b*KV_ + kvh) * L_) * HD_;

  float m_r[8], l_r[8];
  v8f accO[16] = {};
  #pragma unroll
  for (int r = 0; r < 8; ++r){ m_r[r] = -3.0e38f; l_r[r] = 0.f; }
  int segq[8];
  #pragma unroll
  for (int r = 0; r < 8; ++r) segq[r] = seg[b*S_ + q0w + r + 8*hi];

  int lo = q0 - (WIN_ - 1); if (lo < 0) lo = 0;
  const int c0 = lo >> 5, c1 = (q0 + 63) >> 5;

  for (int phase = 0; phase < 2; ++phase){
    const int cbeg = (phase == 0) ? c0 : 0;
    const int cend = (phase == 0) ? c1 : (E_/32 - 1);
    const int base = (phase == 0) ? 0  : S_;
    for (int c = cbeg; c <= cend; ++c){
      const int key0 = base + c*32;
      __syncthreads();
      #pragma unroll
      for (int it = 0; it < 8; ++it){              // stage K, V(transposed)
        int i = threadIdx.x + it*128;
        int row = i >> 5, dq = (i & 31) * 8;
        *(u16x8*)(sK + row*HD_ + dq) =
            *(const u16x8*)(kbase + (size_t)(key0 + row)*HD_ + dq);
        u16x8 vv = *(const u16x8*)(vbase + (size_t)(key0 + row)*HD_ + dq);
        #pragma unroll
        for (int j = 0; j < 8; ++j) sV[(dq + j)*32 + row] = vv[j];
      }
      __syncthreads();
      // ---- scores: Q(16x256) x K^T(256x32) ----
      v8f accS[2] = {};
      #pragma unroll
      for (int ds = 0; ds < 8; ++ds){
        const unsigned short* qp = qbase + (size_t)(q0w + ln)*HD_ + ds*32;
        v8bf a0 = *(const v8bf*)(qp + 8*hi);
        v8bf a1 = *(const v8bf*)(qp + 16 + 8*hi);
        v16bf af = cat8(a0, a1);
        #pragma unroll
        for (int tt = 0; tt < 2; ++tt){
          int key = tt*16 + ln;
          v8bf b0 = *(const v8bf*)(sK + key*HD_ + ds*32 + 8*hi);
          v8bf b1 = *(const v8bf*)(sK + key*HD_ + ds*32 + 16 + 8*hi);
          accS[tt] = wmma_bf16(af, cat8(b0, b1), accS[tt]);
        }
      }
      // ---- scale, softcap, mask ----
      #pragma unroll
      for (int tt = 0; tt < 2; ++tt)
        #pragma unroll
        for (int r = 0; r < 8; ++r){
          float s = accS[tt][r] * 0.0625f;          // 256^-0.5
          s = 50.0f * tanhf(s * 0.02f);             // softcap
          int k = key0 + tt*16 + ln;
          int q = q0w + r + 8*hi;
          bool valid;
          if (phase == 0)
            valid = (k <= q) && (q - k < WIN_) && (decm[b*S_ + k] != 0) &&
                    (seg[b*S_ + k] == segq[r]);
          else
            valid = (encm[b*E_ + (k - S_)] != 0);
          accS[tt][r] = valid ? s : -1.0e30f;
        }
      // ---- online softmax (row reductions within 16-lane halves) ----
      float mc[8];
      #pragma unroll
      for (int r = 0; r < 8; ++r) mc[r] = fmaxf(accS[0][r], accS[1][r]);
      #pragma unroll
      for (int msk = 1; msk <= 8; msk <<= 1)
        #pragma unroll
        for (int r = 0; r < 8; ++r) mc[r] = fmaxf(mc[r], __shfl_xor(mc[r], msk, 32));
      float corr[8];
      #pragma unroll
      for (int r = 0; r < 8; ++r){
        float mn = fmaxf(m_r[r], mc[r]);
        corr[r] = __expf(m_r[r] - mn);
        m_r[r] = mn;
        l_r[r] *= corr[r];
      }
      #pragma unroll
      for (int j = 0; j < 16; ++j)
        #pragma unroll
        for (int r = 0; r < 8; ++r) accO[j][r] *= corr[r];
      float rs[8];
      #pragma unroll
      for (int r = 0; r < 8; ++r) rs[r] = 0.f;
      #pragma unroll
      for (int tt = 0; tt < 2; ++tt)
        #pragma unroll
        for (int r = 0; r < 8; ++r){
          float p = __expf(accS[tt][r] - m_r[r]);
          rs[r] += p;
          sP[w][(r + 8*hi)*32 + tt*16 + ln] = f2bf(p);   // wave-private pane
        }
      #pragma unroll
      for (int msk = 1; msk <= 8; msk <<= 1)
        #pragma unroll
        for (int r = 0; r < 8; ++r) rs[r] += __shfl_xor(rs[r], msk, 32);
      #pragma unroll
      for (int r = 0; r < 8; ++r) l_r[r] += rs[r];
      // ---- O += P(16x32) x V(32x256) ----
      const __bf16* pp = &sP[w][ln*32 + 8*hi];
      v16bf pf = cat8(*(const v8bf*)pp, *(const v8bf*)(pp + 16));
      #pragma unroll
      for (int j = 0; j < 16; ++j){
        int d = j*16 + ln;
        v8bf v0 = *(const v8bf*)(sV + d*32 + 8*hi);
        v8bf v1 = *(const v8bf*)(sV + d*32 + 16 + 8*hi);
        accO[j] = wmma_bf16(pf, cat8(v0, v1), accO[j]);
      }
    }
  }
  // ---- epilogue: normalize, write bf16 (B*S, H*HD) ----
  #pragma unroll
  for (int r = 0; r < 8; ++r){
    float inv = 1.0f / l_r[r];
    int q = q0w + r + 8*hi;
    size_t rowoff = ((size_t)(b*S_ + q))*(H_*HD_) + (size_t)h*HD_;
    #pragma unroll
    for (int j = 0; j < 16; ++j)
      attnout[rowoff + j*16 + ln] = f2bf(accO[j][r] * inv);
  }
}

// -------------------------------------------------------------------------
extern "C" void kernel_launch(void* const* d_in, const int* in_sizes, int n_in,
                              void* d_out, int out_size, void* d_ws, size_t ws_size,
                              hipStream_t stream)
{
  const float* hs   = (const float*)d_in[0];
  const float* ehs  = (const float*)d_in[1];
  const float* cosp = (const float*)d_in[2];
  const float* sinp = (const float*)d_in[3];
  const float* wq   = (const float*)d_in[4];
  const float* wk   = (const float*)d_in[5];
  const float* wv   = (const float*)d_in[6];
  const float* wo   = (const float*)d_in[7];
  const float* qnw  = (const float*)d_in[8];
  const float* knw  = (const float*)d_in[9];
  const int*   decm = (const int*)d_in[10];
  const int*   encm = (const int*)d_in[11];
  const int*   pos  = (const int*)d_in[12];
  float* out = (float*)d_out;

  char* ws = (char*)d_ws;
  size_t off = 0;
  auto take = [&](size_t bytes)->char*{
    char* p = ws + off; off += (bytes + 255) & ~(size_t)255; return p;
  };
  const size_t NTOK = (size_t)B_*S_;   // 2048 tokens (self); same count for encoder
  __bf16* xsb = (__bf16*)take(NTOK*D_*2);
  __bf16* xeb = (__bf16*)take(NTOK*D_*2);
  __bf16* wqb = (__bf16*)take((size_t)D_*(H_*HD_)*2);
  __bf16* wkb = (__bf16*)take((size_t)D_*(KV_*HD_)*2);
  __bf16* wvb = (__bf16*)take((size_t)D_*(KV_*HD_)*2);
  __bf16* wob = (__bf16*)take((size_t)(H_*HD_)*D_*2);
  float*  qf  = (float*) take(NTOK*(H_*HD_)*4);
  float*  ksf = (float*) take(NTOK*(KV_*HD_)*4);
  float*  vsf = (float*) take(NTOK*(KV_*HD_)*4);
  float*  kcf = (float*) take(NTOK*(KV_*HD_)*4);
  float*  vcf = (float*) take(NTOK*(KV_*HD_)*4);
  __bf16* qbb = (__bf16*)take((size_t)B_*H_*S_*HD_*2);
  __bf16* kab = (__bf16*)take((size_t)B_*KV_*L_*HD_*2);
  __bf16* vab = (__bf16*)take((size_t)B_*KV_*L_*HD_*2);
  __bf16* atb = (__bf16*)take(NTOK*(H_*HD_)*2);
  int*    seg = (int*)   take((size_t)B_*S_*4);

  auto cvt = [&](const float* s, __bf16* dptr, size_t n){
    int n4 = (int)(n/4);
    k_cvt_bf16<<<(n4 + 255)/256, 256, 0, stream>>>(s, dptr, n4);
  };
  cvt(hs,  xsb, NTOK*D_);
  cvt(ehs, xeb, NTOK*D_);
  cvt(wq,  wqb, (size_t)D_*H_*HD_);
  cvt(wk,  wkb, (size_t)D_*KV_*HD_);
  cvt(wv,  wvb, (size_t)D_*KV_*HD_);
  cvt(wo,  wob, (size_t)H_*HD_*D_);

  auto gemm = [&](const __bf16* A, const __bf16* Bm, float* C, int M, int N, int K){
    dim3 grid(M/GTM, N/GTN);
    k_gemm_bf16<<<grid, 256, 0, stream>>>((const unsigned short*)A,
                                          (const unsigned short*)Bm, C, M, N, K);
  };
  gemm(xsb, wqb, qf,  B_*S_, H_*HD_,  D_);
  gemm(xsb, wkb, ksf, B_*S_, KV_*HD_, D_);
  gemm(xsb, wvb, vsf, B_*S_, KV_*HD_, D_);
  gemm(xeb, wkb, kcf, B_*E_, KV_*HD_, D_);
  gemm(xeb, wvb, vcf, B_*E_, KV_*HD_, D_);

  k_seg<<<B_, 32, 0, stream>>>(pos, seg);

  // q: norm + rope -> (B,H,S,HD)
  k_norm_rope<<<B_*S_*H_,  HD_, 0, stream>>>(qf,  S_, H_,  qnw, cosp, sinp, qbb, S_, 0);
  // k_self: norm + rope -> kall[:, :, 0:S]
  k_norm_rope<<<B_*S_*KV_, HD_, 0, stream>>>(ksf, S_, KV_, knw, cosp, sinp, kab, L_, 0);
  // k_cross: norm only -> kall[:, :, S:]
  k_norm_rope<<<B_*E_*KV_, HD_, 0, stream>>>(kcf, E_, KV_, knw, nullptr, nullptr, kab, L_, S_);
  // v_self / v_cross: convert + relayout only
  k_norm_rope<<<B_*S_*KV_, HD_, 0, stream>>>(vsf, S_, KV_, nullptr, nullptr, nullptr, vab, L_, 0);
  k_norm_rope<<<B_*E_*KV_, HD_, 0, stream>>>(vcf, E_, KV_, nullptr, nullptr, nullptr, vab, L_, S_);

  k_attn<<<B_*H_*(S_/64), 128, 0, stream>>>(
      (const unsigned short*)qbb, (const unsigned short*)kab,
      (const unsigned short*)vab, seg, decm, encm, atb);

  gemm(atb, wob, out, B_*S_, D_, H_*HD_);

  (void)in_sizes; (void)n_in; (void)out_size; (void)ws_size;
}
